// RelationEncoding_SEG_79860621902602
// MI455X (gfx1250) — matrix operand order repaired and checked by reading, not compiled, not run on hardware
//
#include <hip/hip_runtime.h>
#include <hip/hip_bf16.h>

// RelationEncoding fused kernel set for gfx1250 (MI455X).
// 3 compute passes (global BatchNorm needs grid-wide stats) + 2 tiny finalize
// kernels + 1 zero kernel. MLP GEMMs run on v_wmma_f32_16x16x32_f16.

typedef __attribute__((ext_vector_type(16))) _Float16 v16h;
typedef __attribute__((ext_vector_type(8)))  float    v8f;

#define NPTS   2048
#define NBATCH 8
#define CIN    32
#define NS     32        // nsample
#define R2     64.0f     // radius^2
#define ES     66        // edge LDS row stride (halves), padded: 33 dwords
#define WS     66        // weight LDS row stride
#define HS     66        // h1 LDS row stride
#define CNT_F  524288.0f // B*N*NS values per BN channel
#define BN_EPS 1e-5f

// ---- WMMA fragment loaders (layouts per CDNA5 ISA 7.12.2) ------

// A: 16x32 f16 (M x K). lane<16: rows 0-15, K halves {0..7,16..23};
// lane>=16: same rows, K halves {8..15,24..31}. 2 halves per VGPR.
__device__ __forceinline__ v16h load_A(const _Float16* base, int stride,
                                       int m0, int k0, int lane) {
  const _Float16* row = base + (m0 + (lane & 15)) * stride + k0;
  const int kb = (lane < 16) ? 0 : 8;
  v16h a;
#pragma unroll
  for (int v = 0; v < 4; ++v) {
    a[2 * v]         = row[kb + 2 * v];
    a[2 * v + 1]     = row[kb + 2 * v + 1];
    a[8 + 2 * v]     = row[16 + kb + 2 * v];
    a[8 + 2 * v + 1] = row[16 + kb + 2 * v + 1];
  }
  return a;
}

// B: 32x16 f16 (K x N), from row-major w[o][k] (h[s][o] = sum_k e[s][k]*w[o][k]).
// lane<16: col=lane, K {0..7,16..23}; lane>=16: col=lane-16, K {8..15,24..31}.
__device__ __forceinline__ v16h load_B(const _Float16* w, int stride,
                                       int n0, int k0, int lane) {
  const _Float16* row = w + (n0 + (lane & 15)) * stride + k0 +
                        ((lane < 16) ? 0 : 8);
  v16h b;
#pragma unroll
  for (int j = 0; j < 16; ++j) b[j] = row[(j < 8) ? j : 8 + j];
  return b;
}

__device__ __forceinline__ v8f wmma_f16(v16h a, v16h b, v8f c) {
  return __builtin_amdgcn_wmma_f32_16x16x32_f16(false, a, false, b,
                                                (short)0, c, false, false);
}

// ---- edge tile build: row `lane` = sample, cols [center | nbr-center] -----
__device__ __forceinline__ void build_edge(_Float16* e, const float* pfb,
                                           int n, int m, int lane) {
  _Float16* row = e + lane * ES;
#pragma unroll 8
  for (int c = 0; c < CIN; ++c) {
    float cv = pfb[c * NPTS + n];   // broadcast (same addr all lanes)
    float xv = pfb[c * NPTS + m];   // gather (indices cluster low -> L2 hits)
    row[c]       = (_Float16)cv;
    row[CIN + c] = (_Float16)(xv - cv);
  }
}

// GEMM1 (16x64x64) for one M-tile, apply BN0+leakyReLU, store h1 to LDS f16.
__device__ __forceinline__ void gemm1_bn_h1(const _Float16* ew,
                                            const _Float16* w0h, _Float16* h1w,
                                            const float* sc0, int mtile,
                                            int lane) {
  v16h a0 = load_A(ew, ES, mtile * 16, 0, lane);
  v16h a1 = load_A(ew, ES, mtile * 16, 32, lane);
  const int rbase = (lane < 16) ? 0 : 8;   // C-frag rows for this lane half
#pragma unroll
  for (int nt = 0; nt < 4; ++nt) {
    v8f acc = {};
    acc = wmma_f16(a0, load_B(w0h, WS, nt * 16, 0, lane), acc);
    acc = wmma_f16(a1, load_B(w0h, WS, nt * 16, 32, lane), acc);
    const int ch = nt * 16 + (lane & 15);
    const float sa = sc0[ch], sb = sc0[64 + ch];
#pragma unroll
    for (int v = 0; v < 8; ++v) {
      float x = acc[v] * sa + sb;            // (x-mean)*g/sqrt(var+eps)+be
      x = (x > 0.f) ? x : 0.2f * x;          // leaky ReLU
      h1w[(rbase + v) * HS + ch] = (_Float16)x;
    }
  }
}

// ======================= pass 1: ball query + BN0 stats =====================
__global__ __launch_bounds__(128) void k_pass1(const float* __restrict__ pf,
                                               const float* __restrict__ w0,
                                               int* __restrict__ idxbuf,
                                               float* __restrict__ stats0) {
  __shared__ _Float16 w0h[64 * WS];
  __shared__ _Float16 edge[4][NS * ES];
  __shared__ float    sst[128];
  const int tid = threadIdx.x;
  for (int i = tid; i < 64 * 64; i += 128)
    w0h[(i >> 6) * WS + (i & 63)] = (_Float16)w0[i];
  if (tid < 128) sst[tid] = 0.f;
  __syncthreads();

  const int wave = tid >> 5, lane = tid & 31;
  const int p = blockIdx.x * 4 + wave;       // grid covers B*N exactly
  const int b = p >> 11, n = p & (NPTS - 1);
  const float* pfb = pf + (size_t)b * CIN * NPTS;
  int* nb = idxbuf + (size_t)p * NS;

  // ---- ball query: streaming first-32-qualifying scan, early exit ----
  float cen[CIN];
#pragma unroll
  for (int c = 0; c < CIN; ++c) cen[c] = pfb[c * NPTS + n];
  int count = 0, first = -1, myidx = n;
  for (int base = 0; base < NPTS && count < NS; base += 32) {
    const int m = base + lane;
    float d2 = 0.f;
#pragma unroll
    for (int c = 0; c < CIN; ++c) {
      float t = pfb[c * NPTS + m] - cen[c];
      d2 = fmaf(t, t, d2);
    }
    unsigned mask = __builtin_amdgcn_ballot_w32(d2 <= R2);  // uniform
    while (mask && count < NS) {
      const int bit = __builtin_ctz(mask);
      mask &= mask - 1;
      const int mi = base + bit;
      if (first < 0) first = mi;
      if (lane == count) myidx = mi;         // lane k keeps k-th neighbor
      ++count;
    }
  }
  if (lane >= count) myidx = (first >= 0) ? first : n;  // pad with first hit
  nb[lane] = myidx;                                     // one coalesced store

  build_edge(edge[wave], pfb, n, myidx, lane);
  __syncthreads();  // cross-lane LDS visibility (uniform for all waves)

  // ---- GEMM1 raw (bias cancels under BN) + per-channel sum/sumsq ----
#pragma unroll
  for (int mtile = 0; mtile < 2; ++mtile) {
    v16h a0 = load_A(edge[wave], ES, mtile * 16, 0, lane);
    v16h a1 = load_A(edge[wave], ES, mtile * 16, 32, lane);
#pragma unroll
    for (int nt = 0; nt < 4; ++nt) {
      v8f acc = {};
      acc = wmma_f16(a0, load_B(w0h, WS, nt * 16, 0, lane), acc);
      acc = wmma_f16(a1, load_B(w0h, WS, nt * 16, 32, lane), acc);
      float ls = 0.f, lq = 0.f;
#pragma unroll
      for (int v = 0; v < 8; ++v) { ls += acc[v]; lq += acc[v] * acc[v]; }
      const int ch = nt * 16 + (lane & 15);  // C-frag col = out channel
      atomicAdd(&sst[ch], ls);               // ds_add_f32
      atomicAdd(&sst[64 + ch], lq);
    }
  }
  __syncthreads();
  if (tid < 128) atomicAdd(&stats0[tid], sst[tid]);  // 1 global atomic/ch/blk
}

// ======================= pass 2: GEMM1+BN0 -> GEMM2 raw -> BN1 stats ========
__global__ __launch_bounds__(128) void k_pass2(const float* __restrict__ pf,
                                               const float* __restrict__ w0,
                                               const float* __restrict__ w1,
                                               const int* __restrict__ idxbuf,
                                               const float* __restrict__ sc0,
                                               float* __restrict__ stats1) {
  __shared__ _Float16 w0h[64 * WS];
  __shared__ _Float16 w1h[128 * WS];
  __shared__ _Float16 edge[4][NS * ES];
  __shared__ _Float16 h1[4][16 * HS];
  __shared__ float    sst[256];
  const int tid = threadIdx.x;
  for (int i = tid; i < 64 * 64; i += 128)
    w0h[(i >> 6) * WS + (i & 63)] = (_Float16)w0[i];
  for (int i = tid; i < 128 * 64; i += 128)
    w1h[(i >> 6) * WS + (i & 63)] = (_Float16)w1[i];
  if (tid < 128) { sst[tid] = 0.f; sst[128 + tid] = 0.f; }
  __syncthreads();

  const int wave = tid >> 5, lane = tid & 31;
  const int p = blockIdx.x * 4 + wave;
  const int b = p >> 11, n = p & (NPTS - 1);
  const float* pfb = pf + (size_t)b * CIN * NPTS;
  const int m = idxbuf[(size_t)p * NS + lane];

  build_edge(edge[wave], pfb, n, m, lane);
  __syncthreads();

#pragma unroll
  for (int mtile = 0; mtile < 2; ++mtile) {
    gemm1_bn_h1(edge[wave], w0h, h1[wave], sc0, mtile, lane);
    __syncthreads();  // h1 cross-lane visibility (uniform)
    v16h ha0 = load_A(h1[wave], HS, 0, 0, lane);
    v16h ha1 = load_A(h1[wave], HS, 0, 32, lane);
#pragma unroll
    for (int nt = 0; nt < 8; ++nt) {
      v8f acc = {};
      acc = wmma_f16(ha0, load_B(w1h, WS, nt * 16, 0, lane), acc);
      acc = wmma_f16(ha1, load_B(w1h, WS, nt * 16, 32, lane), acc);
      float ls = 0.f, lq = 0.f;
#pragma unroll
      for (int v = 0; v < 8; ++v) { ls += acc[v]; lq += acc[v] * acc[v]; }
      const int ch = nt * 16 + (lane & 15);
      atomicAdd(&sst[ch], ls);
      atomicAdd(&sst[128 + ch], lq);
    }
    __syncthreads();  // WAR: next mtile overwrites h1
  }
  if (tid < 128) {
    atomicAdd(&stats1[tid], sst[tid]);
    atomicAdd(&stats1[128 + tid], sst[128 + tid]);
  }
}

// ======================= pass 3: full pipeline + max + output ===============
__global__ __launch_bounds__(128) void k_pass3(const float* __restrict__ pf,
                                               const float* __restrict__ w0,
                                               const float* __restrict__ w1,
                                               const int* __restrict__ idxbuf,
                                               const float* __restrict__ sc0,
                                               const float* __restrict__ sc1,
                                               float* __restrict__ out) {
  __shared__ _Float16 w0h[64 * WS];
  __shared__ _Float16 w1h[128 * WS];
  __shared__ _Float16 edge[4][NS * ES];
  __shared__ _Float16 h1[4][16 * HS];
  const int tid = threadIdx.x;
  for (int i = tid; i < 64 * 64; i += 128)
    w0h[(i >> 6) * WS + (i & 63)] = (_Float16)w0[i];
  for (int i = tid; i < 128 * 64; i += 128)
    w1h[(i >> 6) * WS + (i & 63)] = (_Float16)w1[i];
  __syncthreads();

  const int wave = tid >> 5, lane = tid & 31;
  const int p = blockIdx.x * 4 + wave;
  const int b = p >> 11, n = p & (NPTS - 1);
  const float* pfb = pf + (size_t)b * CIN * NPTS;
  const int m = idxbuf[(size_t)p * NS + lane];

  build_edge(edge[wave], pfb, n, m, lane);
  __syncthreads();

  float omax[8];
#pragma unroll
  for (int nt = 0; nt < 8; ++nt) omax[nt] = -3.0e38f;

#pragma unroll
  for (int mtile = 0; mtile < 2; ++mtile) {
    gemm1_bn_h1(edge[wave], w0h, h1[wave], sc0, mtile, lane);
    __syncthreads();
    v16h ha0 = load_A(h1[wave], HS, 0, 0, lane);
    v16h ha1 = load_A(h1[wave], HS, 0, 32, lane);
#pragma unroll
    for (int nt = 0; nt < 8; ++nt) {
      v8f acc = {};
      acc = wmma_f16(ha0, load_B(w1h, WS, nt * 16, 0, lane), acc);
      acc = wmma_f16(ha1, load_B(w1h, WS, nt * 16, 32, lane), acc);
      const int ch = nt * 16 + (lane & 15);
      const float sa = sc1[ch], sb = sc1[128 + ch];
      float mx = -3.0e38f;
#pragma unroll
      for (int v = 0; v < 8; ++v) {      // max over this lane's 8 rows
        float x = acc[v] * sa + sb;
        x = (x > 0.f) ? x : 0.2f * x;
        mx = fmaxf(mx, x);
      }
      mx = fmaxf(mx, __shfl_xor(mx, 16, 32));  // rows 0-7 <-> 8-15
      omax[nt] = fmaxf(omax[nt], mx);          // across sample M-tiles
    }
    __syncthreads();
  }

  if (lane < 16) {
#pragma unroll
    for (int nt = 0; nt < 8; ++nt)
      out[((size_t)b * 128 + nt * 16 + lane) * NPTS + n] = omax[nt];
  }
}

// ======================= small helper kernels ===============================
__global__ void k_zero(float* stats0, float* stats1) {
  const int t = threadIdx.x;
  if (t < 128) stats0[t] = 0.f;
  if (t < 256) stats1[t] = 0.f;
}

// stats -> (scale, shift): xh = x*scale + shift  (bias term folded into mean)
__global__ void k_fin(const float* __restrict__ stats,
                      const float* __restrict__ g,
                      const float* __restrict__ be,
                      float* __restrict__ sc, int C) {
  const int t = threadIdx.x;
  if (t < C) {
    const float mean = stats[t] / CNT_F;
    float var = stats[C + t] / CNT_F - mean * mean;
    var = var < 0.f ? 0.f : var;
    const float a = g[t] * rsqrtf(var + BN_EPS);
    sc[t]     = a;
    sc[C + t] = be[t] - mean * a;
  }
}

extern "C" void kernel_launch(void* const* d_in, const int* in_sizes, int n_in,
                              void* d_out, int out_size, void* d_ws,
                              size_t ws_size, hipStream_t stream) {
  (void)in_sizes; (void)n_in; (void)out_size; (void)ws_size;
  // inputs: points(unused), points_ft, w0, b0(cancels), g0, be0, w1, b1, g1, be1
  const float* pf  = (const float*)d_in[1];
  const float* w0  = (const float*)d_in[2];
  const float* g0  = (const float*)d_in[4];
  const float* be0 = (const float*)d_in[5];
  const float* w1  = (const float*)d_in[6];
  const float* g1  = (const float*)d_in[8];
  const float* be1 = (const float*)d_in[9];
  float* out = (float*)d_out;

  char* ws = (char*)d_ws;                         // needs ~2.1 MB
  int*   idxbuf = (int*)ws;                       // B*N*NS ints = 2 MB
  float* stats0 = (float*)(ws + (size_t)NBATCH * NPTS * NS * 4);
  float* sc0    = stats0 + 128;
  float* stats1 = sc0 + 128;
  float* sc1    = stats1 + 256;

  const int nblk = (NBATCH * NPTS) / 4;           // 4 waves/block, 1 point/wave
  k_zero <<<1, 256, 0, stream>>>(stats0, stats1);
  k_pass1<<<nblk, 128, 0, stream>>>(pf, w0, idxbuf, stats0);
  k_fin  <<<1, 64, 0, stream>>>(stats0, g0, be0, sc0, 64);
  k_pass2<<<nblk, 128, 0, stream>>>(pf, w0, w1, idxbuf, sc0, stats1);
  k_fin  <<<1, 128, 0, stream>>>(stats1, g1, be1, sc1, 128);
  k_pass3<<<nblk, 128, 0, stream>>>(pf, w0, w1, idxbuf, sc0, sc1, out);
}
